// CausalSelfAttention_49323404427872
// MI455X (gfx1250) — matrix operand order repaired and checked by reading, not compiled, not run on hardware
//
#include <hip/hip_runtime.h>

typedef __attribute__((ext_vector_type(16))) __bf16 v16bf;
typedef __attribute__((ext_vector_type(8)))  __bf16 v8bf;
typedef __attribute__((ext_vector_type(8)))  float  v8f;

#define B_N   4
#define T_N   2048
#define H_N   16
#define HD_N  128
#define DM_N  2048
#define MROWS (B_N * T_N)   // 8192
#define LDSTR 40            // padded LDS row stride (elements) -> conflict-free ds_load_b128

union AFrag { v16bf v; v8bf h[2]; };

__device__ __forceinline__ float fast_exp2(float x) {
#if __has_builtin(__builtin_amdgcn_exp2f)
  return __builtin_amdgcn_exp2f(x);   // v_exp_f32 (native exp2)
#else
  return exp2f(x);
#endif
}

__device__ __forceinline__ void wait_async0() {
#if __has_builtin(__builtin_amdgcn_s_wait_asynccnt)
  __builtin_amdgcn_s_wait_asynccnt(0);
#else
  asm volatile("s_wait_asynccnt 0" ::: "memory");
#endif
}

__device__ __forceinline__ void async_copy_b128(unsigned lds_byte_addr,
                                                unsigned long long gaddr) {
  // GLOBAL_LOAD_ASYNC_TO_LDS_B128 (GV mode): VDST = LDS byte address, VADDR = 64-bit global
  asm volatile("global_load_async_to_lds_b128 %0, %1, off"
               :: "v"(lds_byte_addr), "v"(gaddr) : "memory");
}

// ---------------- fp32 -> bf16 conversion ----------------
__global__ __launch_bounds__(256) void k_f32_to_bf16(const float* __restrict__ in,
                                                     void* __restrict__ outv, int n) {
  __bf16* out = (__bf16*)outv;
  int i = blockIdx.x * 256 + threadIdx.x;
  if (i < n) out[i] = (__bf16)in[i];
}

// W [K][N] fp32 -> Wt [N][K] bf16 (so WMMA B-fragments are contiguous loads)
__global__ __launch_bounds__(256) void k_transpose_bf16(const float* __restrict__ in,
                                                        void* __restrict__ outv, int K, int N) {
  __bf16* out = (__bf16*)outv;
  int i = blockIdx.x * 256 + threadIdx.x;
  if (i < K * N) {
    int k = i / N, n = i % N;
    out[(size_t)n * K + k] = (__bf16)in[i];
  }
}

// one k-step: 1 A-fragment load + 8 LDS B-fragments -> 8 back-to-back WMMAs
__device__ __forceinline__ void gemm_step(const __bf16* __restrict__ arow,
                                          const __bf16* Bcur, int k0,
                                          int m, int half, v8f acc[8]) {
  AFrag a;
  a.h[0] = *(const v8bf*)(arow + k0 + 16 * half);
  a.h[1] = *(const v8bf*)(arow + k0 + 16 * half + 8);
  AFrag b[8];
#pragma unroll
  for (int c = 0; c < 8; ++c) {
    const __bf16* bp = Bcur + (c * 16 + m) * LDSTR + 16 * half;
    b[c].h[0] = *(const v8bf*)bp;
    b[c].h[1] = *(const v8bf*)(bp + 8);
  }
#pragma unroll
  for (int c = 0; c < 8; ++c)
    acc[c] = __builtin_amdgcn_wmma_f32_16x16x32_bf16(false, a.v, false, b[c].v,
                                                     (short)0, acc[c], false, false);
}

// ---------------- block-tiled WMMA GEMM (128x128 per block) ----------------
// A [M][K] bf16 row-major, Bt [N][K] bf16 (B transposed).
// 8 waves; wave w: rows [blockRow*128 + w*16), cols [blockCol*128, +128).
// 128x32 B-panel async-staged to double-buffered LDS (latency hidden behind compute).
// mode 0: bf16 -> [B][H][T][HD]; mode 1: bf16 -> [B][H][HD][T]; mode 2: f32 -> [M][N]
__global__ __launch_bounds__(256) void k_gemm_wmma(
    const void* __restrict__ Av, const void* __restrict__ Btv,
    const float* __restrict__ bias, void* __restrict__ out,
    int M, int N, int K, int mode) {
  const __bf16* A  = (const __bf16*)Av;
  const __bf16* Bt = (const __bf16*)Btv;

  __shared__ __bf16 Bl[2][128 * LDSTR];

  int tid  = threadIdx.x;
  int lane = tid & 31;
  int wave = tid >> 5;
  int m    = lane & 15;
  int half = lane >> 4;

  int nbc  = N >> 7;
  int brow = blockIdx.x / nbc;
  int bcol = blockIdx.x % nbc;
  int row0 = (brow << 7) + (wave << 4);
  int colB = bcol << 7;

  const __bf16* arow = A + (size_t)(row0 + m) * K;

  v8f zero = {};
  v8f acc[8];
#pragma unroll
  for (int c = 0; c < 8; ++c) acc[c] = zero;

  // staging: 512 chunks of 8 bf16 (16B); 2 per thread. chunk q: row=q>>2, seg=q&3
  int r0s = tid >> 2,         s0s = tid & 3;
  int r1s = (tid + 256) >> 2, s1s = (tid + 256) & 3;
  unsigned ldsA0 = (unsigned)(unsigned long long)&Bl[0][r0s * LDSTR + s0s * 8];
  unsigned ldsA1 = (unsigned)(unsigned long long)&Bl[0][r1s * LDSTR + s1s * 8];
  unsigned ldsB0 = (unsigned)(unsigned long long)&Bl[1][r0s * LDSTR + s0s * 8];
  unsigned ldsB1 = (unsigned)(unsigned long long)&Bl[1][r1s * LDSTR + s1s * 8];
  const __bf16* g0 = Bt + (size_t)(colB + r0s) * K + s0s * 8;
  const __bf16* g1 = Bt + (size_t)(colB + r1s) * K + s1s * 8;

  // prologue: stage tile 0 into buffer 0
  async_copy_b128(ldsA0, (unsigned long long)g0);
  async_copy_b128(ldsA1, (unsigned long long)g1);

  for (int k0 = 0; k0 < K; k0 += 64) {
    // phase 0: consume buf0 (tile k0), prefetch buf1 (tile k0+32; always in range)
    wait_async0();
    __syncthreads();
    async_copy_b128(ldsB0, (unsigned long long)(g0 + k0 + 32));
    async_copy_b128(ldsB1, (unsigned long long)(g1 + k0 + 32));
    gemm_step(arow, &Bl[0][0], k0, m, half, acc);

    // phase 1: consume buf1 (tile k0+32), prefetch buf0 (tile k0+64) if it exists
    wait_async0();
    __syncthreads();
    if (k0 + 64 < K) {
      async_copy_b128(ldsA0, (unsigned long long)(g0 + k0 + 64));
      async_copy_b128(ldsA1, (unsigned long long)(g1 + k0 + 64));
    }
    gemm_step(arow, &Bl[1][0], k0 + 32, m, half, acc);
  }

#pragma unroll
  for (int c = 0; c < 8; ++c) {
    int gcol = colB + c * 16 + m;
    float bv = bias ? bias[gcol] : 0.0f;
#pragma unroll
    for (int r = 0; r < 8; ++r) {
      int grow = row0 + r + 8 * half;
      float val = acc[c][r] + bv;
      if (mode == 2) {
        ((float*)out)[(size_t)grow * N + gcol] = val;
      } else {
        int bb = grow / T_N, t = grow % T_N;
        int h = gcol >> 7,  d = gcol & 127;
        size_t idx = (mode == 0)
          ? ((size_t)(bb * H_N + h) * T_N + t) * HD_N + d
          : ((size_t)(bb * H_N + h) * HD_N + d) * T_N + t;
        ((__bf16*)out)[idx] = (__bf16)val;
      }
    }
  }
}

// ---------------- flash attention: one wave per 32 query rows ----------------
__device__ __forceinline__ float red_max16(float x) {
#pragma unroll
  for (int off = 8; off >= 1; off >>= 1) x = fmaxf(x, __shfl_xor(x, off, 32));
  return x;
}
__device__ __forceinline__ float red_sum16(float x) {
#pragma unroll
  for (int off = 8; off >= 1; off >>= 1) x += __shfl_xor(x, off, 32);
  return x;
}

__global__ __launch_bounds__(32) void k_flash_attn(
    const void* __restrict__ Qv, const void* __restrict__ Kv,
    const void* __restrict__ Vtv, void* __restrict__ Outv) {
  const __bf16* Q  = (const __bf16*)Qv;    // [BH][T][128]
  const __bf16* Km = (const __bf16*)Kv;    // [BH][T][128]
  const __bf16* Vt = (const __bf16*)Vtv;   // [BH][128][T]
  __bf16* Out = (__bf16*)Outv;             // [B][T][H*128]

  __shared__ __bf16 ldsP[16][32];

  int lane = threadIdx.x & 31;
  int m    = lane & 15;
  int half = lane >> 4;

  int ntq = T_N >> 5;                 // 64 tiles of 32 query rows
  int bh  = blockIdx.x / ntq;
  int it  = blockIdx.x % ntq;
  int i0  = it << 5;

  // Q fragments for two 16-row groups (K-fragments/V-fragments shared by both)
  AFrag qf[2][4];
#pragma unroll
  for (int g = 0; g < 2; ++g) {
    const __bf16* qrow = Q + ((size_t)bh * T_N + i0 + 16 * g + m) * HD_N;
#pragma unroll
    for (int kk = 0; kk < 4; ++kk) {
      qf[g][kk].h[0] = *(const v8bf*)(qrow + kk * 32 + 16 * half);
      qf[g][kk].h[1] = *(const v8bf*)(qrow + kk * 32 + 16 * half + 8);
    }
  }

  v8f zero = {};
  v8f o[2][8];
  float mrun[2][8], lrun[2][8];
#pragma unroll
  for (int g = 0; g < 2; ++g) {
#pragma unroll
    for (int dt = 0; dt < 8; ++dt) o[g][dt] = zero;
#pragma unroll
    for (int r = 0; r < 8; ++r) { mrun[g][r] = -3e38f; lrun[g][r] = 0.0f; }
  }

  // exp2-domain softmax: fold 1/sqrt(128) * log2(e) into the logit scale
  const float scale2 = 0.088388347648318447f * 1.4426950408889634f;

  for (int j0 = 0; j0 < i0 + 32; j0 += 32) {
    // S = Q*K^T for both groups, sharing K fragments
    v8f s00 = zero, s01 = zero, s10 = zero, s11 = zero;
    const __bf16* krow0 = Km + ((size_t)bh * T_N + j0 + m) * HD_N;
    const __bf16* krow1 = krow0 + (size_t)16 * HD_N;
#pragma unroll
    for (int kk = 0; kk < 4; ++kk) {
      v16bf b0 = *(const v16bf*)(krow0 + kk * 32 + 16 * half);
      v16bf b1 = *(const v16bf*)(krow1 + kk * 32 + 16 * half);
      s00 = __builtin_amdgcn_wmma_f32_16x16x32_bf16(false, qf[0][kk].v, false, b0, (short)0, s00, false, false);
      s01 = __builtin_amdgcn_wmma_f32_16x16x32_bf16(false, qf[0][kk].v, false, b1, (short)0, s01, false, false);
      s10 = __builtin_amdgcn_wmma_f32_16x16x32_bf16(false, qf[1][kk].v, false, b0, (short)0, s10, false, false);
      s11 = __builtin_amdgcn_wmma_f32_16x16x32_bf16(false, qf[1][kk].v, false, b1, (short)0, s11, false, false);
    }
    // V fragments once, reused by both groups
    AFrag vf[8];
#pragma unroll
    for (int dt = 0; dt < 8; ++dt)
      vf[dt].v = *(const v16bf*)(Vt + ((size_t)bh * HD_N + dt * 16 + m) * T_N + j0 + 16 * half);

#pragma unroll
    for (int g = 0; g < 2; ++g) {
      v8f& sa = g ? s10 : s00;
      v8f& sb = g ? s11 : s01;
      int ig = i0 + 16 * g;
      bool need_mask = (j0 + 31 >= ig);
#pragma unroll
      for (int r = 0; r < 8; ++r) {
        sa[r] *= scale2; sb[r] *= scale2;
        if (need_mask) {
          int irow = ig + r + 8 * half;
          if (j0 + m > irow)      sa[r] = -3e38f;
          if (j0 + 16 + m > irow) sb[r] = -3e38f;
        }
      }
#pragma unroll
      for (int r = 0; r < 8; ++r) {
        float rowm  = red_max16(fmaxf(sa[r], sb[r]));
        float newm  = fmaxf(mrun[g][r], rowm);
        float alpha = fast_exp2(mrun[g][r] - newm);
        mrun[g][r] = newm;
        float p0 = fast_exp2(sa[r] - newm);
        float p1 = fast_exp2(sb[r] - newm);
        sa[r] = p0; sb[r] = p1;
        lrun[g][r] = lrun[g][r] * alpha + red_sum16(p0 + p1);
#pragma unroll
        for (int dt = 0; dt < 8; ++dt) o[g][dt][r] *= alpha;
      }
      // transpose P (C/D layout) -> A-fragment layout via LDS
#pragma unroll
      for (int r = 0; r < 8; ++r) {
        ldsP[r + 8 * half][m]      = (__bf16)sa[r];
        ldsP[r + 8 * half][m + 16] = (__bf16)sb[r];
      }
      __syncthreads();
      AFrag pf;
      pf.h[0] = *(const v8bf*)&ldsP[m][16 * half];
      pf.h[1] = *(const v8bf*)&ldsP[m][16 * half + 8];
      __syncthreads();
#pragma unroll
      for (int dt = 0; dt < 8; ++dt)
        o[g][dt] = __builtin_amdgcn_wmma_f32_16x16x32_bf16(false, pf.v, false, vf[dt].v,
                                                           (short)0, o[g][dt], false, false);
    }
  }

  int bb = bh / H_N, h = bh % H_N;
#pragma unroll
  for (int g = 0; g < 2; ++g) {
#pragma unroll
    for (int r = 0; r < 8; ++r) {
      float inv = 1.0f / lrun[g][r];
      int t = i0 + 16 * g + r + 8 * half;
      __bf16* orow = Out + ((size_t)bb * T_N + t) * DM_N + h * HD_N;
#pragma unroll
      for (int dt = 0; dt < 8; ++dt) orow[dt * 16 + m] = (__bf16)(o[g][dt][r] * inv);
    }
  }
}

extern "C" void kernel_launch(void* const* d_in, const int* in_sizes, int n_in,
                              void* d_out, int out_size, void* d_ws, size_t ws_size,
                              hipStream_t stream) {
  const float* query = (const float*)d_in[0];
  const float* w_q   = (const float*)d_in[1];
  const float* b_q   = (const float*)d_in[2];
  const float* w_k   = (const float*)d_in[3];
  const float* b_k   = (const float*)d_in[4];
  const float* w_v   = (const float*)d_in[5];
  const float* b_v   = (const float*)d_in[6];
  const float* w_o   = (const float*)d_in[7];
  const float* b_o   = (const float*)d_in[8];

  char* p = (char*)d_ws;
  const size_t XBYTES = (size_t)MROWS * DM_N * 2;  // 32 MB
  const size_t WBYTES = (size_t)DM_N * DM_N * 2;   //  8 MB
  void* Xbf = p; p += XBYTES;
  void* Wqt = p; p += WBYTES;
  void* Wkt = p; p += WBYTES;
  void* Wvt = p; p += WBYTES;
  void* Wot = p; p += WBYTES;
  void* Qb  = p; p += XBYTES;
  void* Kb  = p; p += XBYTES;
  void* Vtb = p; p += XBYTES;
  void* Att = p; p += XBYTES;   // total ~192 MB

  int nX = MROWS * DM_N;
  k_f32_to_bf16<<<nX / 256, 256, 0, stream>>>(query, Xbf, nX);
  int nW = DM_N * DM_N;
  k_transpose_bf16<<<nW / 256, 256, 0, stream>>>(w_q, Wqt, DM_N, DM_N);
  k_transpose_bf16<<<nW / 256, 256, 0, stream>>>(w_k, Wkt, DM_N, DM_N);
  k_transpose_bf16<<<nW / 256, 256, 0, stream>>>(w_v, Wvt, DM_N, DM_N);
  k_transpose_bf16<<<nW / 256, 256, 0, stream>>>(w_o, Wot, DM_N, DM_N);

  int blocks = (MROWS / 128) * (DM_N / 128);  // 1024 blocks of 256
  k_gemm_wmma<<<blocks, 256, 0, stream>>>(Xbf, Wqt, b_q, Qb,  MROWS, DM_N, DM_N, 0);
  k_gemm_wmma<<<blocks, 256, 0, stream>>>(Xbf, Wkt, b_k, Kb,  MROWS, DM_N, DM_N, 0);
  k_gemm_wmma<<<blocks, 256, 0, stream>>>(Xbf, Wvt, b_v, Vtb, MROWS, DM_N, DM_N, 1);

  k_flash_attn<<<B_N * H_N * (T_N / 32), 32, 0, stream>>>(Qb, Kb, Vtb, Att);

  k_gemm_wmma<<<blocks, 256, 0, stream>>>(Att, Wot, b_o, d_out, MROWS, DM_N, DM_N, 2);
}
